// HybridLineSampling_85444079386726
// MI455X (gfx1250) — compile-verified, matrix operand order checked by reading, not run on hardware
//
#include <hip/hip_runtime.h>

// ---------------- problem constants ----------------
#define NLINES   8192
#define CCH      256
#define HH       128
#define WW       128
#define NTOK     17          // 8 line groups + 9 region offsets
#define K1       (NTOK*CCH)  // 4352
#define HID      512
#define OUTC     256

// ---------------- WMMA types ----------------
typedef __attribute__((ext_vector_type(16))) __bf16 bf16x16;
typedef __attribute__((ext_vector_type(8)))  float  f32x8;

union AFrag { bf16x16 v; uint4 q[2]; unsigned u32[8]; unsigned short u16[16]; };
union CFrag { f32x8  v; float f[8]; };

__device__ __forceinline__ unsigned short f2bf(float f) {
    unsigned u = __builtin_bit_cast(unsigned, f);
    unsigned r = u + 0x7FFFu + ((u >> 16) & 1u);
    return (unsigned short)(r >> 16);
}

// Async global->LDS copy (CDNA5). lds ptr low 32 bits == LDS byte offset.
__device__ __forceinline__ void async_copy_b128(const void* gptr, void* lptr) {
    unsigned lds = (unsigned)(size_t)lptr;
    unsigned long long ga = (unsigned long long)(size_t)gptr;
    asm volatile("global_load_async_to_lds_b128 %0, %1, off"
                 :: "v"(lds), "v"(ga) : "memory");
}
__device__ __forceinline__ void async_copy_b32(const void* gptr, void* lptr) {
    unsigned lds = (unsigned)(size_t)lptr;
    unsigned long long ga = (unsigned long long)(size_t)gptr;
    asm volatile("global_load_async_to_lds_b32 %0, %1, off"
                 :: "v"(lds), "v"(ga) : "memory");
}
// Wait until at most N async ops outstanding (in-order completion -> drains older stage).
template <int N>
__device__ __forceinline__ void async_wait_le() {
    asm volatile("s_wait_asynccnt %0" :: "n"(N) : "memory");
}

// Load one 16x16x32 bf16 A/B fragment from LDS laid out row-major (row = M or N).
// ISA layout: lane L holds row L%16; VGPR r holds K pair (r>=4?16:0)+8*(L>>4)+2*(r&3)
// -> two contiguous 16B runs at half-offsets 8*kh and 16+8*kh.
__device__ __forceinline__ void load_frag(AFrag& f, const unsigned short* base,
                                          int row, int ld, int kh) {
    f.q[0] = *(const uint4*)&base[row * ld + 8 * kh];
    f.q[1] = *(const uint4*)&base[row * ld + 16 + 8 * kh];
}

// ---------------- fp32 -> bf16 transposed weight conversion ----------------
// src: (K, N) row-major f32 ; dst: (N, K) row-major bf16
__global__ __launch_bounds__(256) void cvt_transpose_kernel(
    const float* __restrict__ src, unsigned short* __restrict__ dst, int K, int N)
{
    int k = blockIdx.x * 256 + threadIdx.x;
    int n = blockIdx.y;
    if (k < K) dst[(size_t)n * K + k] = f2bf(src[(size_t)k * N + n]);
}

// ---------------- sampling: one block per line, one lane per channel ----------------
__global__ __launch_bounds__(256) void sample_kernel(
    const float* __restrict__ fm,      // (B, C, H, W)
    const float* __restrict__ starts,  // (N, 2)
    const float* __restrict__ ends,    // (N, 2)
    const int*   __restrict__ bidx,    // (N,)
    const int*   __restrict__ imgp,    // scalar
    unsigned short* __restrict__ tok)  // (N, 17*256) bf16
{
    const int n = blockIdx.x;
    const int c = threadIdx.x;
    const float img  = (float)imgp[0];
    const float rimg = 1.0f / img;
    const float sx = starts[2*n + 0], sy = starts[2*n + 1];
    const float ex = ends[2*n + 0],   ey = ends[2*n + 1];
    const int   b  = bidx[n];
    const float* __restrict__ fmc = fm + ((size_t)(b * CCH + c)) * (HH * WW);

    auto samp = [&](float gx, float gy) -> float {
        float x = fminf(fmaxf((gx + 1.0f) * 0.5f * (float)(WW - 1), 0.0f), (float)(WW - 1));
        float y = fminf(fmaxf((gy + 1.0f) * 0.5f * (float)(HH - 1), 0.0f), (float)(HH - 1));
        float x0f = floorf(x), y0f = floorf(y);
        int x0 = (int)x0f, y0 = (int)y0f;
        int x1 = min(x0 + 1, WW - 1), y1 = min(y0 + 1, HH - 1);
        float wx = x - x0f, wy = y - y0f;
        float f00 = fmc[y0 * WW + x0];
        float f01 = fmc[y0 * WW + x1];
        float f10 = fmc[y1 * WW + x0];
        float f11 = fmc[y1 * WW + x1];
        float top = f00 + (f01 - f00) * wx;
        float bot = f10 + (f11 - f10) * wx;
        return top + (bot - top) * wy;
    };

    unsigned short* __restrict__ trow = tok + (size_t)n * K1;
    const float dx = ex - sx, dy = ey - sy;

    #pragma unroll
    for (int g = 0; g < 8; ++g) {
        float mv = -3.4e38f;
        #pragma unroll
        for (int j = 0; j < 4; ++j) {
            int i = g * 4 + j;
            float t  = (float)i * (1.0f / 31.0f);
            float px = sx + t * dx, py = sy + t * dy;
            mv = fmaxf(mv, samp(px * rimg * 2.0f - 1.0f, py * rimg * 2.0f - 1.0f));
        }
        trow[g * CCH + c] = f2bf(mv);
    }

    const float cx = (sx + ex) * 0.5f, cy = (sy + ey) * 0.5f;
    float len  = fmaxf(sqrtf(dx * dx + dy * dy), 20.0f);
    float half = len * 0.5f * rimg;
    float cnx = cx * rimg * 2.0f - 1.0f;
    float cny = cy * rimg * 2.0f - 1.0f;
    const float offx[9] = {0.f,-1.f,1.f, 0.f,0.f,-1.f, 1.f,-1.f,1.f};
    const float offy[9] = {0.f, 0.f,0.f,-1.f,1.f,-1.f,-1.f, 1.f,1.f};
    #pragma unroll
    for (int j = 0; j < 9; ++j) {
        trow[(8 + j) * CCH + c] = f2bf(samp(cnx + offx[j] * half, cny + offy[j] * half));
    }
}

// ---------------- GEMM1: tokens(8192x4352) @ W1(4352x512) + b1, GELU -> bf16 ----------------
// Block tile 128x128, 8 waves in 4x2 grid, wave tile 32x64, BK=32, double-buffered async.
#define LDA1 40   // 32 + 8 pad (half units)
#define LDB1 40

__global__ __launch_bounds__(256) void gemm1_kernel(
    const unsigned short* __restrict__ A,    // tokens bf16 (8192 x 4352)
    const unsigned short* __restrict__ BwT,  // W1^T bf16 (512 x 4352)
    const float* __restrict__ b1,
    unsigned short* __restrict__ Hout)       // (8192, 512) bf16
{
    __shared__ alignas(16) unsigned short sA[2][128 * LDA1];   // M x K
    __shared__ alignas(16) unsigned short sB[2][128 * LDB1];   // N x K (transposed)

    const int tid  = threadIdx.x;
    const int lane = tid & 31;
    const int wave = tid >> 5;
    const int wm = wave & 3, wn = wave >> 2;     // 4x2 -> wave tile 32(M) x 64(N)
    const int bm = blockIdx.x * 128;
    const int bn = blockIdx.y * 128;
    const int f16i = lane & 15;
    const int kh   = lane >> 4;

    CFrag acc[2][4];
    #pragma unroll
    for (int i = 0; i < 2; ++i)
        #pragma unroll
        for (int j = 0; j < 4; ++j)
            #pragma unroll
            for (int r = 0; r < 8; ++r) acc[i][j].f[r] = 0.0f;

    // 4 async b128 ops per thread per stage (2 for A, 2 for B)
    auto issue = [&](int ki, int buf) {
        const int kk = ki * 32;
        #pragma unroll
        for (int q = 0; q < 2; ++q) {
            int chunk = tid + q * 256;                 // 512 chunks: 128 rows x 4
            int row = chunk >> 2, ch = (chunk & 3) * 8;
            async_copy_b128(&A[(size_t)(bm + row) * K1 + kk + ch],
                            &sA[buf][row * LDA1 + ch]);
        }
        #pragma unroll
        for (int q = 0; q < 2; ++q) {
            int chunk = tid + q * 256;                 // 512 chunks: 128 n-rows x 4
            int n = chunk >> 2, ch = (chunk & 3) * 8;
            async_copy_b128(&BwT[(size_t)(bn + n) * K1 + kk + ch],
                            &sB[buf][n * LDB1 + ch]);
        }
    };

    auto compute = [&](int buf) {
        AFrag af[2], bf[4];
        #pragma unroll
        for (int t = 0; t < 2; ++t)
            load_frag(af[t], sA[buf], wm * 32 + t * 16 + f16i, LDA1, kh);
        #pragma unroll
        for (int t = 0; t < 4; ++t)
            load_frag(bf[t], sB[buf], wn * 64 + t * 16 + f16i, LDB1, kh);
        #pragma unroll
        for (int i = 0; i < 2; ++i)
            #pragma unroll
            for (int j = 0; j < 4; ++j)
                acc[i][j].v = __builtin_amdgcn_wmma_f32_16x16x32_bf16(
                    false, af[i].v, false, bf[j].v, (short)0, acc[i][j].v, false, false);
    };

    const int NK = K1 / 32;   // 136
    issue(0, 0);
    for (int i = 0; i < NK - 1; ++i) {
        issue(i + 1, (i + 1) & 1);     // prefetch next stage into other buffer
        async_wait_le<4>();            // drain previous stage (in-order)
        __syncthreads();
        compute(i & 1);
        __syncthreads();               // reads done before this buffer is refilled
    }
    async_wait_le<0>();
    __syncthreads();
    compute((NK - 1) & 1);

    // epilogue: bias + exact GELU -> bf16
    #pragma unroll
    for (int i = 0; i < 2; ++i) {
        #pragma unroll
        for (int j = 0; j < 4; ++j) {
            int col  = bn + wn * 64 + j * 16 + f16i;
            float bias = b1[col];
            #pragma unroll
            for (int r = 0; r < 8; ++r) {
                int row = bm + wm * 32 + i * 16 + r + 8 * kh;
                float x = acc[i][j].f[r] + bias;
                float g = 0.5f * x * (1.0f + erff(x * 0.70710678118f));
                Hout[(size_t)row * HID + col] = f2bf(g);
            }
        }
    }
}

// ---------------- GEMM2: H(8192x512) @ W2(512x256) + b2, fused LayerNorm ----------------
#define LDA2 40
#define LDB2 40

__global__ __launch_bounds__(256) void gemm2_kernel(
    const unsigned short* __restrict__ Hb,   // bf16 (8192 x 512)
    const unsigned short* __restrict__ BwT,  // W2^T bf16 (256 x 512)
    const float* __restrict__ b2,
    const float* __restrict__ gamma,
    const float* __restrict__ beta,
    float* __restrict__ out)                 // (8192, 256) f32
{
    __shared__ alignas(16) unsigned short sA[2][16 * LDA2];    // M x K
    __shared__ alignas(16) unsigned short sB[2][256 * LDB2];   // N x K (transposed)
    __shared__ float sO[16 * OUTC];

    const int tid  = threadIdx.x;
    const int lane = tid & 31;
    const int wave = tid >> 5;               // 8 waves -> 32 cols each
    const int bm   = blockIdx.x * 16;
    const int f16i = lane & 15;
    const int kh   = lane >> 4;

    CFrag acc[2];
    #pragma unroll
    for (int t = 0; t < 2; ++t)
        #pragma unroll
        for (int r = 0; r < 8; ++r) acc[t].f[r] = 0.0f;

    // 5 async ops per thread per stage (1 b32 for A, 4 b128 for B)
    auto issue = [&](int ki, int buf) {
        const int kk = ki * 32;
        int row = tid >> 4, ch2 = (tid & 15) * 2;
        async_copy_b32(&Hb[(size_t)(bm + row) * HID + kk + ch2],
                       &sA[buf][row * LDA2 + ch2]);
        #pragma unroll
        for (int q = 0; q < 4; ++q) {              // 1024 chunks: 256 n-rows x 4
            int chunk = tid + q * 256;
            int n = chunk >> 2, ch = (chunk & 3) * 8;
            async_copy_b128(&BwT[(size_t)n * HID + kk + ch],
                            &sB[buf][n * LDB2 + ch]);
        }
    };

    auto compute = [&](int buf) {
        AFrag af;
        load_frag(af, sA[buf], f16i, LDA2, kh);
        #pragma unroll
        for (int t = 0; t < 2; ++t) {
            AFrag bf;
            load_frag(bf, sB[buf], wave * 32 + t * 16 + f16i, LDB2, kh);
            acc[t].v = __builtin_amdgcn_wmma_f32_16x16x32_bf16(
                false, af.v, false, bf.v, (short)0, acc[t].v, false, false);
        }
    };

    const int NK = HID / 32;   // 16
    issue(0, 0);
    for (int i = 0; i < NK - 1; ++i) {
        issue(i + 1, (i + 1) & 1);
        async_wait_le<5>();
        __syncthreads();
        compute(i & 1);
        __syncthreads();
    }
    async_wait_le<0>();
    __syncthreads();
    compute((NK - 1) & 1);

    // stage outputs (+bias) in LDS for row-wise LayerNorm
    #pragma unroll
    for (int t = 0; t < 2; ++t) {
        int col  = wave * 32 + t * 16 + f16i;
        float bias = b2[col];
        #pragma unroll
        for (int r = 0; r < 8; ++r)
            sO[(r + 8 * kh) * OUTC + col] = acc[t].f[r] + bias;
    }
    __syncthreads();

    // LayerNorm: 16 threads per row, 16 cols per thread, xor-reduce over 16 lanes
    int row = tid >> 4;
    int sub = tid & 15;
    float s = 0.0f, ss = 0.0f;
    #pragma unroll
    for (int j = 0; j < 16; ++j) {
        float v = sO[row * OUTC + sub + 16 * j];
        s += v; ss += v * v;
    }
    #pragma unroll
    for (int m = 8; m >= 1; m >>= 1) {
        s  += __shfl_xor(s,  m, 16);
        ss += __shfl_xor(ss, m, 16);
    }
    float mu  = s * (1.0f / 256.0f);
    float var = ss * (1.0f / 256.0f) - mu * mu;
    float inv = rsqrtf(var + 1e-5f);
    #pragma unroll
    for (int j = 0; j < 16; ++j) {
        int c = sub + 16 * j;
        float v = sO[row * OUTC + c];
        out[(size_t)(bm + row) * OUTC + c] = (v - mu) * inv * gamma[c] + beta[c];
    }
}

// ---------------- launcher ----------------
extern "C" void kernel_launch(void* const* d_in, const int* in_sizes, int n_in,
                              void* d_out, int out_size, void* d_ws, size_t ws_size,
                              hipStream_t stream) {
    const float* fm     = (const float*)d_in[0];
    const float* starts = (const float*)d_in[1];
    const float* ends   = (const float*)d_in[2];
    const int*   bidx   = (const int*)d_in[3];
    const int*   imgp   = (const int*)d_in[4];
    const float* W1     = (const float*)d_in[5];
    const float* b1     = (const float*)d_in[6];
    const float* W2     = (const float*)d_in[7];
    const float* b2     = (const float*)d_in[8];
    const float* gamma  = (const float*)d_in[9];
    const float* beta   = (const float*)d_in[10];
    float* out = (float*)d_out;

    // workspace layout (bf16 halves), all 16B-aligned section sizes
    unsigned short* tok  = (unsigned short*)d_ws;             // 8192*4352
    unsigned short* w1t  = tok  + (size_t)NLINES * K1;        // 512*4352  (W1^T)
    unsigned short* w2t  = w1t  + (size_t)HID * K1;           // 256*512   (W2^T)
    unsigned short* hb   = w2t  + (size_t)OUTC * HID;         // 8192*512

    cvt_transpose_kernel<<<dim3(K1 / 256, HID), 256, 0, stream>>>(W1, w1t, K1, HID);
    cvt_transpose_kernel<<<dim3(HID / 256, OUTC), 256, 0, stream>>>(W2, w2t, HID, OUTC);
    sample_kernel<<<NLINES, 256, 0, stream>>>(fm, starts, ends, bidx, imgp, tok);
    gemm1_kernel<<<dim3(NLINES / 128, HID / 128), 256, 0, stream>>>(tok, w1t, b1, hb);
    gemm2_kernel<<<NLINES / 16, 256, 0, stream>>>(hb, w2t, b2, gamma, beta, out);
}
